// Deformable_Encoder_26018911879858
// MI455X (gfx1250) — matrix operand order repaired and checked by reading, not compiled, not run on hardware
//
#include <hip/hip_runtime.h>
#include <math.h>

// ---------------------------------------------------------------------------
// Types for CDNA5 WMMA (wave32, v_wmma_f32_16x16x32_bf16)
// ---------------------------------------------------------------------------
typedef __bf16 bf16;
typedef __attribute__((ext_vector_type(16))) __bf16 v16bf;
typedef __attribute__((ext_vector_type(8)))  __bf16 v8bf;
typedef __attribute__((ext_vector_type(8)))  float  v8f;

union ABFrag {
    v16bf v;
    struct { v8bf lo, hi; } h;
};

__device__ __forceinline__ bf16 f2bf(float f) { return (bf16)f; }

// Problem constants (from reference: N=4, Lq=6400, C=256, H=W=80, heads=8, pts=4)
#define NB    4
#define LQ    6400
#define CC    256
#define HH    80
#define WW    80
#define MROWS (NB * LQ)      // 25600
#define HID   1024
#define NPAD_OFFA 128        // 64 offsets + 32 attn logits, padded to 128

// ---------------------------------------------------------------------------
// f32 -> bf16 convert (vectorized by 4)
// ---------------------------------------------------------------------------
__global__ __launch_bounds__(256) void k_cvt_bf16(const float* __restrict__ in,
                                                  bf16* __restrict__ out, int n4) {
    int i = blockIdx.x * 256 + threadIdx.x;
    if (i >= n4) return;
    float4 v = ((const float4*)in)[i];
    struct alignas(8) B4 { bf16 a, b, c, d; };
    B4 o; o.a = f2bf(v.x); o.b = f2bf(v.y); o.c = f2bf(v.z); o.d = f2bf(v.w);
    ((B4*)out)[i] = o;
}

// ---------------------------------------------------------------------------
// Weight transpose + convert: W (K x N, row-major f32) -> Bt (N x K, bf16)
// ---------------------------------------------------------------------------
__global__ __launch_bounds__(256) void k_transpose_bf16(const float* __restrict__ Win,
                                                        bf16* __restrict__ out,
                                                        int K, int N) {
    int i = blockIdx.x * 256 + threadIdx.x;
    if (i >= K * N) return;
    int n = i / K;
    int k = i - n * K;
    out[i] = f2bf(Win[(size_t)k * N + n]);
}

// Build fused (Woff | Wa | zero-pad) transposed B: 128 rows x 256 cols
__global__ __launch_bounds__(256) void k_build_offa_bt(const float* __restrict__ Woff,
                                                       const float* __restrict__ Wa,
                                                       bf16* __restrict__ out) {
    int i = blockIdx.x * 256 + threadIdx.x;   // over 128*256
    if (i >= NPAD_OFFA * CC) return;
    int n = i / CC;
    int k = i - n * CC;
    float v;
    if (n < 64)       v = Woff[(size_t)k * 64 + n];
    else if (n < 96)  v = Wa[(size_t)k * 32 + (n - 64)];
    else              v = 0.0f;
    out[i] = f2bf(v);
}

__global__ void k_bias128(const float* __restrict__ boff, const float* __restrict__ ba,
                          float* __restrict__ out) {
    int i = threadIdx.x;   // 128 threads
    float v = 0.0f;
    if (i < 64)      v = boff[i];
    else if (i < 96) v = ba[i - 64];
    out[i] = v;
}

// ---------------------------------------------------------------------------
// LayerNorm over C=256, one wave (32 lanes) per row, 8 elems/lane -> bf16 out
// ---------------------------------------------------------------------------
__global__ __launch_bounds__(256) void k_layernorm_bf16(const float* __restrict__ x,
                                                        const float* __restrict__ g,
                                                        const float* __restrict__ b,
                                                        bf16* __restrict__ out) {
    int row  = blockIdx.x * 8 + (threadIdx.x >> 5);
    int lane = threadIdx.x & 31;
    const float* p = x + (size_t)row * CC + lane * 8;
    float4 u0 = *(const float4*)p;
    float4 u1 = *(const float4*)(p + 4);
    float e[8] = {u0.x, u0.y, u0.z, u0.w, u1.x, u1.y, u1.z, u1.w};

    float s = 0.f, ss = 0.f;
#pragma unroll
    for (int k = 0; k < 8; ++k) { s += e[k]; ss += e[k] * e[k]; }
#pragma unroll
    for (int off = 16; off > 0; off >>= 1) {
        s  += __shfl_xor(s,  off, 32);
        ss += __shfl_xor(ss, off, 32);
    }
    float mean = s * (1.0f / CC);
    float var  = ss * (1.0f / CC) - mean * mean;
    float rstd = rsqrtf(var + 1e-5f);

    ABFrag frag;  // only use lo half for 8 bf16 outputs
    union { v8bf v; bf16 e[8]; } o;
#pragma unroll
    for (int k = 0; k < 8; ++k) {
        int c = lane * 8 + k;
        o.e[k] = f2bf((e[k] - mean) * rstd * g[c] + b[c]);
    }
    *(v8bf*)(out + (size_t)row * CC + lane * 8) = o.v;
    (void)frag;
}

// ---------------------------------------------------------------------------
// Tiled bf16 WMMA GEMM: C[M,N] = A[M,K] (bf16) * Bt[N,K]^T (bf16) + bias
// 256 threads = 8 waves; block tile 128x128; wave tile 32x64 (2x4 WMMA frags)
// epi: 0 = store f32, 2 = store f32 + residual, 3 = GELU -> bf16
// ---------------------------------------------------------------------------
__global__ __launch_bounds__(256) void k_gemm_bf16(const bf16* __restrict__ A,
                                                   const bf16* __restrict__ Bt,
                                                   const float* __restrict__ bias,
                                                   const float* __restrict__ residual,
                                                   float* __restrict__ outF,
                                                   bf16* __restrict__ outB,
                                                   int M, int N, int K, int epi) {
    const int tid   = threadIdx.x;
    const int wave  = tid >> 5;
    const int lane  = tid & 31;
    const int lrow  = lane & 15;   // M (A-frag) / N (B-frag) / N (C-frag)
    const int khalf = lane >> 4;   // K-half selector

    const int nTilesN = N >> 7;
    const int bM = (blockIdx.x / nTilesN) * 128;
    const int bN = (blockIdx.x % nTilesN) * 128;
    const int wm = (wave & 3) << 5;   // 0,32,64,96
    const int wn = (wave >> 2) << 6;  // 0,64

    v8f zero = {0.f, 0.f, 0.f, 0.f, 0.f, 0.f, 0.f, 0.f};
    v8f acc[2][4];
#pragma unroll
    for (int i = 0; i < 2; ++i)
#pragma unroll
        for (int j = 0; j < 4; ++j) acc[i][j] = zero;

    // A fragment base: lane holds row (bM+wm+i*16+lrow), K chunks khalf*8+{0..7} and +16
    const bf16* aBase = A + (size_t)(bM + wm + lrow) * K + khalf * 8;
    // B fragment base: lane holds col (bN+wn+j*16+lrow), K chunks khalf*8+{0..7} and +16
    const bf16* bBase = Bt + (size_t)(bN + wn + lrow) * K + khalf * 8;

    for (int k0 = 0; k0 < K; k0 += 32) {
        ABFrag a[2], bq[4];
#pragma unroll
        for (int i = 0; i < 2; ++i) {
            const bf16* p = aBase + (size_t)i * 16 * K + k0;
            a[i].h.lo = *(const v8bf*)(p);
            a[i].h.hi = *(const v8bf*)(p + 16);
        }
#pragma unroll
        for (int j = 0; j < 4; ++j) {
            const bf16* p = bBase + (size_t)j * 16 * K + k0;
            bq[j].h.lo = *(const v8bf*)(p);
            bq[j].h.hi = *(const v8bf*)(p + 16);
        }
#pragma unroll
        for (int i = 0; i < 2; ++i)
#pragma unroll
            for (int j = 0; j < 4; ++j)
                acc[i][j] = __builtin_amdgcn_wmma_f32_16x16x32_bf16(
                    false, a[i].v, false, bq[j].v, (short)0, acc[i][j], false, false);
    }

    // Epilogue. C/D layout: col = lrow, vgpr r -> row = r + 8*khalf
#pragma unroll
    for (int i = 0; i < 2; ++i) {
#pragma unroll
        for (int j = 0; j < 4; ++j) {
            int n  = bN + wn + j * 16 + lrow;
            float bvv = bias[n];
            int mBase = bM + wm + i * 16 + khalf * 8;
#pragma unroll
            for (int r = 0; r < 8; ++r) {
                size_t idx = (size_t)(mBase + r) * N + n;
                float v = acc[i][j][r] + bvv;
                if (epi == 0) {
                    outF[idx] = v;
                } else if (epi == 2) {
                    outF[idx] = v + residual[idx];
                } else {  // epi == 3 : exact GELU -> bf16
                    float gelu = 0.5f * v * (1.0f + erff(v * 0.70710678118654752f));
                    outB[idx] = f2bf(gelu);
                }
            }
        }
    }
}

// ---------------------------------------------------------------------------
// Deformable sampling: one wave per (n, q, head); lane = channel d (hd=32).
// offa row layout: [0..63] offsets (h*8 + p*2 + {x,y}), [64..95] attn logits.
// vbuf layout: (n, pos, c) f32 so each corner gather is a coalesced 128B read.
// ---------------------------------------------------------------------------
__global__ __launch_bounds__(256) void k_sample(const float* __restrict__ offa,
                                                const float* __restrict__ ref,
                                                const float* __restrict__ vbuf,
                                                bf16* __restrict__ out) {
    int gid  = blockIdx.x * 8 + (threadIdx.x >> 5);
    int lane = threadIdx.x & 31;
    int h = gid & 7;
    int q = (gid >> 3) % LQ;
    int n = gid / (8 * LQ);

    size_t qbase = (size_t)n * LQ + q;
    const float* oa = offa + qbase * NPAD_OFFA;
    float rx = ref[qbase * 2 + 0];
    float ry = ref[qbase * 2 + 1];

    // softmax over 4 point logits
    float l0 = oa[64 + h * 4 + 0], l1 = oa[64 + h * 4 + 1];
    float l2 = oa[64 + h * 4 + 2], l3 = oa[64 + h * 4 + 3];
    float mx = fmaxf(fmaxf(l0, l1), fmaxf(l2, l3));
    float e0 = expf(l0 - mx), e1 = expf(l1 - mx), e2 = expf(l2 - mx), e3 = expf(l3 - mx);
    float inv = 1.0f / (e0 + e1 + e2 + e3);
    float aw[4] = {e0 * inv, e1 * inv, e2 * inv, e3 * inv};

    const float* vb = vbuf + (size_t)n * LQ * CC + h * 32 + lane;
    auto corner = [&](int xi, int yi) -> float {
        if ((unsigned)xi < (unsigned)WW && (unsigned)yi < (unsigned)HH)
            return vb[(size_t)(yi * WW + xi) * CC];
        return 0.0f;
    };

    float accv = 0.0f;
#pragma unroll
    for (int p = 0; p < 4; ++p) {
        float ox = oa[h * 8 + p * 2 + 0];
        float oy = oa[h * 8 + p * 2 + 1];
        // loc = ref + off/normalizer; pixel = loc*size - 0.5  ==  ref*size + off - 0.5
        float gx = rx * WW + ox - 0.5f;
        float gy = ry * HH + oy - 0.5f;
        float x0f = floorf(gx), y0f = floorf(gy);
        float wx = gx - x0f, wy = gy - y0f;
        int x0 = (int)x0f, y0 = (int)y0f;
        float s = corner(x0,     y0)     * ((1.f - wx) * (1.f - wy))
                + corner(x0 + 1, y0)     * (wx * (1.f - wy))
                + corner(x0,     y0 + 1) * ((1.f - wx) * wy)
                + corner(x0 + 1, y0 + 1) * (wx * wy);
        accv += aw[p] * s;
    }
    out[qbase * CC + h * 32 + lane] = f2bf(accv);
}

// ---------------------------------------------------------------------------
// Host-side launch
// ---------------------------------------------------------------------------
extern "C" void kernel_launch(void* const* d_in, const int* in_sizes, int n_in,
                              void* d_out, int out_size, void* d_ws, size_t ws_size,
                              hipStream_t stream) {
    const float* x     = (const float*)d_in[0];
    const float* ref   = (const float*)d_in[1];
    const float* value = (const float*)d_in[2];
    const float* ln1_w = (const float*)d_in[5];
    const float* ln1_b = (const float*)d_in[6];
    const float* ln2_w = (const float*)d_in[7];
    const float* ln2_b = (const float*)d_in[8];
    const float* Wv    = (const float*)d_in[9];
    const float* bv    = (const float*)d_in[10];
    const float* Woff  = (const float*)d_in[11];
    const float* boff  = (const float*)d_in[12];
    const float* Wa    = (const float*)d_in[13];
    const float* ba    = (const float*)d_in[14];
    const float* Wout  = (const float*)d_in[15];
    const float* bout  = (const float*)d_in[16];
    const float* W1    = (const float*)d_in[17];
    const float* b1    = (const float*)d_in[18];
    const float* W2    = (const float*)d_in[19];
    const float* b2    = (const float*)d_in[20];
    float* out = (float*)d_out;

    // Workspace layout (256B aligned regions)
    char* w = (char*)d_ws;
    auto alloc = [&](size_t bytes) -> char* {
        char* p = w;
        w += (bytes + 255) & ~(size_t)255;
        return p;
    };
    bf16*  WvT     = (bf16*)alloc((size_t)CC * CC * 2);
    bf16*  BtOffa  = (bf16*)alloc((size_t)NPAD_OFFA * CC * 2);
    bf16*  WoutT   = (bf16*)alloc((size_t)CC * CC * 2);
    bf16*  W1T     = (bf16*)alloc((size_t)HID * CC * 2);
    bf16*  W2T     = (bf16*)alloc((size_t)CC * HID * 2);
    float* bias128 = (float*)alloc(NPAD_OFFA * 4);
    bf16*  val_bf  = (bf16*)alloc((size_t)MROWS * CC * 2);
    bf16*  ln1_bf  = (bf16*)alloc((size_t)MROWS * CC * 2);
    float* vbuf    = (float*)alloc((size_t)MROWS * CC * 4);
    float* offa    = (float*)alloc((size_t)MROWS * NPAD_OFFA * 4);
    float* x1      = (float*)alloc((size_t)MROWS * CC * 4);
    bf16*  h1_bf   = (bf16*)alloc((size_t)MROWS * HID * 2);
    bf16*  samp_bf = ln1_bf;   // ln1 consumed by offa GEMM before sampling
    bf16*  hln_bf  = val_bf;   // value_bf consumed by v-proj GEMM before LN2

    // ---- weight prep (one-time per call; tiny) ----
    k_transpose_bf16<<<(CC * CC + 255) / 256, 256, 0, stream>>>(Wv, WvT, CC, CC);
    k_build_offa_bt<<<(NPAD_OFFA * CC + 255) / 256, 256, 0, stream>>>(Woff, Wa, BtOffa);
    k_transpose_bf16<<<(CC * CC + 255) / 256, 256, 0, stream>>>(Wout, WoutT, CC, CC);
    k_transpose_bf16<<<(CC * HID + 255) / 256, 256, 0, stream>>>(W1, W1T, CC, HID);
    k_transpose_bf16<<<(HID * CC + 255) / 256, 256, 0, stream>>>(W2, W2T, HID, CC);
    k_bias128<<<1, 128, 0, stream>>>(boff, ba, bias128);
    k_cvt_bf16<<<(MROWS * CC / 4 + 255) / 256, 256, 0, stream>>>(value, val_bf, MROWS * CC / 4);

    // ---- ln1(x) -> bf16 ----
    k_layernorm_bf16<<<MROWS / 8, 256, 0, stream>>>(x, ln1_w, ln1_b, ln1_bf);

    // ---- value projection: vbuf = value @ Wv + bv (f32) ----
    k_gemm_bf16<<<(MROWS / 128) * (CC / 128), 256, 0, stream>>>(
        val_bf, WvT, bv, nullptr, vbuf, nullptr, MROWS, CC, CC, 0);

    // ---- offsets + attention logits: offa = ln1 @ [Woff|Wa] + [boff|ba] ----
    k_gemm_bf16<<<(MROWS / 128) * (NPAD_OFFA / 128), 256, 0, stream>>>(
        ln1_bf, BtOffa, bias128, nullptr, offa, nullptr, MROWS, NPAD_OFFA, CC, 0);

    // ---- deformable bilinear sampling + softmax-weighted sum -> bf16 ----
    k_sample<<<(NB * LQ * 8) / 8, 256, 0, stream>>>(offa, ref, vbuf, samp_bf);

    // ---- out projection + residual: x1 = x + samp @ Wout + bout ----
    k_gemm_bf16<<<(MROWS / 128) * (CC / 128), 256, 0, stream>>>(
        samp_bf, WoutT, bout, x, x1, nullptr, MROWS, CC, CC, 2);

    // ---- ln2(x1) -> bf16 ----
    k_layernorm_bf16<<<MROWS / 8, 256, 0, stream>>>(x1, ln2_w, ln2_b, hln_bf);

    // ---- FC1 + exact GELU -> bf16 ----
    k_gemm_bf16<<<(MROWS / 128) * (HID / 128), 256, 0, stream>>>(
        hln_bf, W1T, b1, nullptr, nullptr, h1_bf, MROWS, HID, CC, 3);

    // ---- FC2 + residual -> final f32 output ----
    k_gemm_bf16<<<(MROWS / 128) * (CC / 128), 256, 0, stream>>>(
        h1_bf, W2T, b2, x1, out, nullptr, MROWS, CC, HID, 2);

    (void)in_sizes; (void)n_in; (void)out_size; (void)ws_size;
}
